// CellBlock_17703855194354
// MI455X (gfx1250) — matrix operand order repaired and checked by reading, not compiled, not run on hardware
//
#include <hip/hip_runtime.h>
#include <hip/hip_bf16.h>
#include <math.h>

#define N_NODES 100000
#define N_CELLS 200000
#define N_EDGES 300000
#define D 128
#define INV_SCALE 0.08838834764831845f  // 1/sqrt(128)

typedef __attribute__((ext_vector_type(16))) _Float16 v16h;
typedef __attribute__((ext_vector_type(4)))  _Float16 v4h;
typedef __attribute__((ext_vector_type(8)))  float    v8f;
typedef __attribute__((ext_vector_type(4)))  float    v4f;

// ---------------- workspace layout (bytes) ----------------
constexpr size_t alignup(size_t x) { return (x + 255) & ~(size_t)255; }
constexpr size_t OFF_DOTR = 0;                                           // E f32
constexpr size_t OFF_DOTS = alignup(OFF_DOTR + (size_t)N_EDGES * 4);     // E f32
constexpr size_t OFF_NMAX = alignup(OFF_DOTS + (size_t)N_EDGES * 4);     // N u32 (flipped f32)
constexpr size_t OFF_NSUM = alignup(OFF_NMAX + (size_t)N_NODES * 4);     // N f32
constexpr size_t OFF_NCNT = alignup(OFF_NSUM + (size_t)N_NODES * 4);     // N f32
constexpr size_t OFF_AGG  = alignup(OFF_NCNT + (size_t)N_NODES * 4);     // N*128 f32
constexpr size_t OFF_XH   = alignup(OFF_AGG + (size_t)N_NODES * D * 4);  // C*256 f16
constexpr size_t OFF_W1P  = alignup(OFF_XH + (size_t)N_CELLS * 2 * D * 2); // 256*128 f16 packed
constexpr size_t OFF_W2P  = alignup(OFF_W1P + (size_t)256 * 128 * 2);      // 128*128 f16 packed
constexpr size_t WS_TOTAL = alignup(OFF_W2P + (size_t)128 * 128 * 2);

// ---------------- helpers ----------------
__device__ __forceinline__ unsigned fflip(float f) {
  unsigned u = __float_as_uint(f);
  return (u & 0x80000000u) ? ~u : (u | 0x80000000u);
}
__device__ __forceinline__ float funflip(unsigned u) {
  return __uint_as_float((u & 0x80000000u) ? (u ^ 0x80000000u) : ~u);
}

// ---------------- 0) init + weight fragment packing ----------------
// B fragment layout for v_wmma_f32_16x16x32_f16 (per 16x16 N-tile, 32-K chunk):
//   lane 0..15 : n = lane,     K = 0..7, 16..23
//   lane 16..31: n = lane-16,  K = 8..15, 24..31
// Packed as [kchunk][ntile][lane][16 halves] -> contiguous 32B per lane.
__global__ void k_init(float* __restrict__ agg, float* __restrict__ outNode,
                       unsigned* __restrict__ nmax, float* __restrict__ nsum,
                       float* __restrict__ ncnt,
                       const float* __restrict__ W1, const float* __restrict__ W2,
                       _Float16* __restrict__ W1p, _Float16* __restrict__ W2p) {
  int i = blockIdx.x * blockDim.x + threadIdx.x;
  int stride = gridDim.x * blockDim.x;
  for (int j = i; j < N_NODES * D; j += stride) { agg[j] = 0.f; outNode[j] = 0.f; }
  for (int j = i; j < N_NODES; j += stride) { nmax[j] = 0u; nsum[j] = 0.f; ncnt[j] = 0.f; }
  for (int j = i; j < 256 * 128; j += stride) {
    int h = j & 15, lane = (j >> 4) & 31, nt = (j >> 9) & 7, kc = j >> 12;
    int K = kc * 32 + ((lane >> 4) << 3) + (h & 7) + ((h >> 3) << 4);
    int n = nt * 16 + (lane & 15);
    W1p[j] = (_Float16)W1[K * 128 + n];
    if (j < 128 * 128) {
      int kc2 = j >> 12;  // 0..3
      int K2 = kc2 * 32 + ((lane >> 4) << 3) + (h & 7) + ((h >> 3) << 4);
      W2p[j] = (_Float16)W2[K2 * 128 + n];
    }
  }
}

// ---------------- 1) per-edge attention dots + segment max ----------------
__global__ __launch_bounds__(256) void k_att_dot(const float* __restrict__ edge_attr,
                                                 const float* __restrict__ node_emb,
                                                 const int* __restrict__ ei,
                                                 float* __restrict__ dotR, float* __restrict__ dotS,
                                                 unsigned* __restrict__ nmax) {
  const int wave = threadIdx.x >> 5, lane = threadIdx.x & 31;
  const int e = blockIdx.x * 8 + wave;
  if (e >= N_EDGES) return;
  const int s = ei[e];             // senders = edge_index[0]
  const int r = ei[N_EDGES + e];   // receivers = edge_index[1]
  const v4f ea = *(const v4f*)(edge_attr + (size_t)e * D + lane * 4);
  const v4f nr = *(const v4f*)(node_emb + (size_t)r * D + lane * 4);
  const v4f ns = *(const v4f*)(node_emb + (size_t)s * D + lane * 4);
  float dr = ea.x * nr.x + ea.y * nr.y + ea.z * nr.z + ea.w * nr.w;
  float ds = ea.x * ns.x + ea.y * ns.y + ea.z * ns.z + ea.w * ns.w;
  #pragma unroll
  for (int off = 16; off; off >>= 1) {
    dr += __shfl_xor(dr, off, 32);
    ds += __shfl_xor(ds, off, 32);
  }
  if (lane == 0) {
    dr *= INV_SCALE; ds *= INV_SCALE;
    dotR[e] = dr; dotS[e] = ds;
    atomicMax(&nmax[r], fflip(dr));
    atomicMax(&nmax[s], fflip(ds));
  }
}

// ---------------- 2) exp + segment sum ----------------
__global__ void k_expsum(const int* __restrict__ ei, const float* __restrict__ dotR,
                         const float* __restrict__ dotS, const unsigned* __restrict__ nmax,
                         float* __restrict__ nsum) {
  int e = blockIdx.x * blockDim.x + threadIdx.x;
  if (e >= N_EDGES) return;
  int s = ei[e], r = ei[N_EDGES + e];
  atomicAdd(&nsum[r], expf(dotR[e] - funflip(nmax[r])));
  atomicAdd(&nsum[s], expf(dotS[e] - funflip(nmax[s])));
}

// ---------------- 3) weighted scatter of edge features ----------------
__global__ __launch_bounds__(256) void k_agg(const float* __restrict__ edge_attr,
                                             const int* __restrict__ ei,
                                             const float* __restrict__ dotR,
                                             const float* __restrict__ dotS,
                                             const unsigned* __restrict__ nmax,
                                             const float* __restrict__ nsum,
                                             float* __restrict__ agg) {
  const int wave = threadIdx.x >> 5, lane = threadIdx.x & 31;
  const int e = blockIdx.x * 8 + wave;
  if (e >= N_EDGES) return;
  const int s = ei[e], r = ei[N_EDGES + e];
  const float wr = expf(dotR[e] - funflip(nmax[r])) / nsum[r];
  const float ws = expf(dotS[e] - funflip(nmax[s])) / nsum[s];
  const v4f ea = *(const v4f*)(edge_attr + (size_t)e * D + lane * 4);
  float* ar = agg + (size_t)r * D + lane * 4;
  float* as = agg + (size_t)s * D + lane * 4;
  atomicAdd(ar + 0, ea.x * wr); atomicAdd(ar + 1, ea.y * wr);
  atomicAdd(ar + 2, ea.z * wr); atomicAdd(ar + 3, ea.w * wr);
  atomicAdd(as + 0, ea.x * ws); atomicAdd(as + 1, ea.y * ws);
  atomicAdd(as + 2, ea.z * ws); atomicAdd(as + 3, ea.w * ws);
}

// ---------------- 4) cell gather + f16 pack of MLP input, node counts ----------------
__global__ __launch_bounds__(256) void k_gather(const float* __restrict__ cell_attr,
                                                const float* __restrict__ agg,
                                                const int* __restrict__ face,
                                                _Float16* __restrict__ Xh,
                                                float* __restrict__ ncnt) {
  const int wave = threadIdx.x >> 5, lane = threadIdx.x & 31;
  const int c = blockIdx.x * 8 + wave;
  if (c >= N_CELLS) return;
  const int f0 = face[c], f1 = face[N_CELLS + c], f2 = face[2 * N_CELLS + c];
  const v4f ca = *(const v4f*)(cell_attr + (size_t)c * D + lane * 4);
  const v4f g0 = *(const v4f*)(agg + (size_t)f0 * D + lane * 4);
  const v4f g1 = *(const v4f*)(agg + (size_t)f1 * D + lane * 4);
  const v4f g2 = *(const v4f*)(agg + (size_t)f2 * D + lane * 4);
  const float third = 1.0f / 3.0f;
  v4h xa, xg;
  xa.x = (_Float16)ca.x; xa.y = (_Float16)ca.y; xa.z = (_Float16)ca.z; xa.w = (_Float16)ca.w;
  xg.x = (_Float16)((g0.x + g1.x + g2.x) * third);
  xg.y = (_Float16)((g0.y + g1.y + g2.y) * third);
  xg.z = (_Float16)((g0.z + g1.z + g2.z) * third);
  xg.w = (_Float16)((g0.w + g1.w + g2.w) * third);
  *(v4h*)(Xh + (size_t)c * 256 + lane * 4) = xa;
  *(v4h*)(Xh + (size_t)c * 256 + 128 + lane * 4) = xg;
  if (lane == 0) {
    atomicAdd(&ncnt[f0], 1.0f);
    atomicAdd(&ncnt[f1], 1.0f);
    atomicAdd(&ncnt[f2], 1.0f);
  }
}

// ---------------- 5) fused 2-layer MLP via WMMA + node scatter ----------------
// Per wave: one 16-row tile of X [16x256] -> h [16x128] -> out [16x128].
// A-fragment layout (16-bit A 16x32): lane holds row M=lane&15;
//   lanes 0-15 hold K {0..7, 16..23}, lanes 16-31 hold K {8..15, 24..31}.
__global__ __launch_bounds__(256) void k_mlp(const _Float16* __restrict__ Xh,
                                             const _Float16* __restrict__ W1p,
                                             const _Float16* __restrict__ W2p,
                                             const float* __restrict__ b1,
                                             const float* __restrict__ b2,
                                             const int* __restrict__ face,
                                             float* __restrict__ outCell,
                                             float* __restrict__ outNode) {
  __shared__ _Float16 hst[8][16][136];  // per-wave relu(h) staging, padded rows
  const int wave = threadIdx.x >> 5, lane = threadIdx.x & 31;
  const int tile = blockIdx.x * 8 + wave;
  if (tile >= N_CELLS / 16) return;  // no block-wide sync below; early exit is safe
  const int row0 = tile * 16;
  const int hi = lane >> 4;   // 0 = low lane half, 1 = high
  const int n0 = lane & 15;
  const v8f vzero = {0.f, 0.f, 0.f, 0.f, 0.f, 0.f, 0.f, 0.f};

  v8f acc[8];
  #pragma unroll
  for (int nt = 0; nt < 8; ++nt) acc[nt] = vzero;

  // ---- layer 1: [16x256] @ [256x128] ----
  const _Float16* arow = Xh + (size_t)(row0 + n0) * 256 + hi * 8;
  #pragma unroll
  for (int kc = 0; kc < 8; ++kc) {
    v16h a;
    const _Float16* p = arow + kc * 32;
    #pragma unroll
    for (int h2 = 0; h2 < 8; ++h2) { a[h2] = p[h2]; a[8 + h2] = p[16 + h2]; }
    #pragma unroll
    for (int nt = 0; nt < 8; ++nt) {
      const v16h b = *(const v16h*)(W1p + (((kc << 3) + nt) << 9) + (lane << 4));
      acc[nt] = __builtin_amdgcn_wmma_f32_16x16x32_f16(false, a, false, b,
                                                       (short)0, acc[nt], false, false);
    }
  }

  // ---- bias + relu -> LDS (C/D layout: VGPR r -> M = r + 8*hi, N = n0) ----
  #pragma unroll
  for (int nt = 0; nt < 8; ++nt) {
    const float bb = b1[nt * 16 + n0];
    #pragma unroll
    for (int r = 0; r < 8; ++r) {
      float v = acc[nt][r] + bb;
      v = v > 0.f ? v : 0.f;
      hst[wave][r + hi * 8][nt * 16 + n0] = (_Float16)v;
    }
    acc[nt] = vzero;
  }
  asm volatile("s_wait_dscnt 0" ::: "memory");  // LDS stores visible before re-read

  // ---- layer 2: [16x128] @ [128x128] ----
  const _Float16* hrow = &hst[wave][n0][hi * 8];
  #pragma unroll
  for (int kc = 0; kc < 4; ++kc) {
    v16h a;
    const _Float16* p = hrow + kc * 32;
    #pragma unroll
    for (int h2 = 0; h2 < 8; ++h2) { a[h2] = p[h2]; a[8 + h2] = p[16 + h2]; }
    #pragma unroll
    for (int nt = 0; nt < 8; ++nt) {
      const v16h b = *(const v16h*)(W2p + (((kc << 3) + nt) << 9) + (lane << 4));
      acc[nt] = __builtin_amdgcn_wmma_f32_16x16x32_f16(false, a, false, b,
                                                       (short)0, acc[nt], false, false);
    }
  }

  // ---- epilogue: bias, store cell_attr_new, fused scatter-add to nodes ----
  int f0[8], f1[8], f2[8];
  #pragma unroll
  for (int r = 0; r < 8; ++r) {
    int row = row0 + r + hi * 8;
    f0[r] = face[row]; f1[r] = face[N_CELLS + row]; f2[r] = face[2 * N_CELLS + row];
  }
  #pragma unroll
  for (int nt = 0; nt < 8; ++nt) {
    const float bb = b2[nt * 16 + n0];
    const int n = nt * 16 + n0;
    #pragma unroll
    for (int r = 0; r < 8; ++r) {
      const float v = acc[nt][r] + bb;
      const int row = row0 + r + hi * 8;
      outCell[(size_t)row * D + n] = v;
      atomicAdd(&outNode[(size_t)f0[r] * D + n], v);
      atomicAdd(&outNode[(size_t)f1[r] * D + n], v);
      atomicAdd(&outNode[(size_t)f2[r] * D + n], v);
    }
  }
}

// ---------------- 6) node mean finalize ----------------
__global__ void k_fin(float* __restrict__ outNode, const float* __restrict__ ncnt) {
  int i = blockIdx.x * blockDim.x + threadIdx.x;
  if (i >= N_NODES * D) return;
  outNode[i] /= fmaxf(ncnt[i >> 7], 1.0f);
}

// ---------------- host launcher ----------------
extern "C" void kernel_launch(void* const* d_in, const int* in_sizes, int n_in,
                              void* d_out, int out_size, void* d_ws, size_t ws_size,
                              hipStream_t stream) {
  (void)in_sizes; (void)n_in; (void)out_size;
  if (ws_size < WS_TOTAL) return;

  const float* cell_attr = (const float*)d_in[0];
  const float* edge_attr = (const float*)d_in[1];
  const float* node_emb  = (const float*)d_in[2];
  const int*   edge_idx  = (const int*)d_in[3];
  const int*   face      = (const int*)d_in[4];
  const float* W1 = (const float*)d_in[5];
  const float* b1 = (const float*)d_in[6];
  const float* W2 = (const float*)d_in[7];
  const float* b2 = (const float*)d_in[8];

  float* outCell = (float*)d_out;
  float* outNode = outCell + (size_t)N_CELLS * D;

  char* ws = (char*)d_ws;
  float*    dotR = (float*)(ws + OFF_DOTR);
  float*    dotS = (float*)(ws + OFF_DOTS);
  unsigned* nmax = (unsigned*)(ws + OFF_NMAX);
  float*    nsum = (float*)(ws + OFF_NSUM);
  float*    ncnt = (float*)(ws + OFF_NCNT);
  float*    agg  = (float*)(ws + OFF_AGG);
  _Float16* Xh   = (_Float16*)(ws + OFF_XH);
  _Float16* W1p  = (_Float16*)(ws + OFF_W1P);
  _Float16* W2p  = (_Float16*)(ws + OFF_W2P);

  k_init<<<2048, 256, 0, stream>>>(agg, outNode, nmax, nsum, ncnt, W1, W2, W1p, W2p);
  k_att_dot<<<N_EDGES / 8, 256, 0, stream>>>(edge_attr, node_emb, edge_idx, dotR, dotS, nmax);
  k_expsum<<<(N_EDGES + 255) / 256, 256, 0, stream>>>(edge_idx, dotR, dotS, nmax, nsum);
  k_agg<<<N_EDGES / 8, 256, 0, stream>>>(edge_attr, edge_idx, dotR, dotS, nmax, nsum, agg);
  k_gather<<<N_CELLS / 8, 256, 0, stream>>>(cell_attr, agg, face, Xh, ncnt);
  k_mlp<<<(N_CELLS / 16 + 7) / 8, 256, 0, stream>>>(Xh, W1p, W2p, b1, b2, face, outCell, outNode);
  k_fin<<<(N_NODES * D + 255) / 256, 256, 0, stream>>>(outNode, ncnt);
}
